// TransformerModel_69681549410390
// MI455X (gfx1250) — matrix operand order repaired and checked by reading, not compiled, not run on hardware
//
#include <hip/hip_runtime.h>

// ---- problem constants (match reference) ----
#define S_      2048
#define B_      4
#define IN_     64
#define H_      1024
#define DFF_    1024
#define OUT_    64
#define L_      3
#define NHEAD_  4
#define WINDOW_ 4
#define DH_     (H_ / NHEAD_)   // 256

#define CHUNK_K 256             // K-elements staged per TDM round

typedef __bf16 bf16_t;
typedef __attribute__((ext_vector_type(16))) __bf16 v16bf;
typedef __attribute__((ext_vector_type(8)))  __bf16 v8bf;
typedef __attribute__((ext_vector_type(8)))  float  v8f;
typedef __attribute__((ext_vector_type(4)))  unsigned int u32x4;
typedef __attribute__((ext_vector_type(8)))  unsigned int u32x8;

__device__ __forceinline__ v16bf concat8(v8bf lo, v8bf hi) {
  return __builtin_shufflevector(lo, hi, 0,1,2,3,4,5,6,7,8,9,10,11,12,13,14,15);
}

// ------------------------------------------------------------------
// f32 -> bf16 conversion (weights / src), grid-stride
// ------------------------------------------------------------------
__global__ void k_cvt_bf16(const float* __restrict__ in, bf16_t* __restrict__ out, size_t n) {
  size_t i = (size_t)blockIdx.x * blockDim.x + threadIdx.x;
  size_t stride = (size_t)gridDim.x * blockDim.x;
  for (; i < n; i += stride) out[i] = (bf16_t)in[i];
}

// ------------------------------------------------------------------
// TDM: DMA a 16-row x kcnt*32-col slice of W (row-major, row stride K
// bf16) into LDS, laid out in WMMA B-fragment order:
//   fastest->slowest walk: u(2x8B=16B) | nlow(16 rows) | khalf(2, +16 elems)
//                          | chunkhalf(2, +8 elems)    | k0s(kcnt, +32 elems)
// LDS written contiguously in walk order => consumer lane L reads its
// 16B at frag + L*16 (lo) and frag + 512 + L*16 (hi): conflict-free.
// Descriptor bitfields per CDNA5 ISA 8.3-8.6 (D# groups 0..3), 5-D mode.
// ------------------------------------------------------------------
__device__ __forceinline__ void tdm_stage_B(const bf16_t* gsrc, unsigned lds_off,
                                            int K, int kcnt)
{
  unsigned long long ga = (unsigned long long)(uintptr_t)gsrc;
  u32x4 g0; u32x8 g1; u32x4 g2; u32x4 g3;

  // group0: count=1 | lds_addr | global_addr | type=2
  g0[0] = 1u;
  g0[1] = lds_off;
  g0[2] = (unsigned)(ga & 0xFFFFFFFFu);
  g0[3] = (unsigned)((ga >> 32) & 0x01FFFFFFu) | (2u << 30);

  // group1
  g1[0] = 3u << 16;                       // wg_mask=0, data_size=3 (8B units)
  g1[1] = 2u << 16;                       // tensor_dim0 = 2   (bits 79:48 lo)
  g1[2] = 16u << 16;                      // tensor_dim1 = 16  (bits 111:80 lo)
  g1[3] = 2u << 16;                       // tile_dim0 = 2     (bits 127:112)
  g1[4] = 16u | (2u << 16);               // tile_dim1=16, tile_dim2=2
  g1[5] = (unsigned)(K >> 2);             // dim0_stride = K/4 units (row)
  g1[6] = 4u << 16;                       // dim1_stride = 4 units (khalf: +32B)
  g1[7] = 0u;

  // group2 (normal mode, iterate disabled)
  g2[0] = 2u;                             // tensor_dim2 = 2
  g2[1] = 2u;                             // tensor_dim3 = 2
  g2[2] = 2u;                             // dim2_stride = 2 units (chunkhalf: +16B)
  g2[3] = 2u << 16;                       // tile_dim3 = 2

  // group3
  g3[0] = 8u;                             // dim3_stride = 8 units (k-step: +64B)
  g3[1] = (unsigned)kcnt << 16;           // tensor_dim4 = kcnt
  g3[2] = (unsigned)kcnt << 16;           // tile_dim4 = kcnt
  g3[3] = 0u;

  asm volatile("tensor_load_to_lds %0, %1, %2, %3"
               :: "s"(g0), "s"(g1), "s"(g2), "s"(g3)
               : "memory");
}

// ------------------------------------------------------------------
// C[M,N] = A[M,K] @ W[N,K]^T (+bias) (+relu), bf16 in, f32 accum.
// Block = 256 thr = 8 waves; block tile 256x64; wave tile 32x64
// (2x4 WMMA 16x16x32 bf16 tiles). A streamed from global per wave
// (rows unique per wave); B staged to LDS by TDM, double-buffered,
// in fragment order (conflict-free ds_load_b128 reads).
// ------------------------------------------------------------------
__global__ __launch_bounds__(256) void k_gemm_bf16(
    const bf16_t* __restrict__ A, const bf16_t* __restrict__ W,
    const float* __restrict__ bias, float* __restrict__ Cf,
    bf16_t* __restrict__ Cb, int M, int N, int K, int relu)
{
  // 2 buffers x (4 j-tiles x 8 k-steps x 512 elems) = 2 x 32KB
  __shared__ __align__(128) bf16_t Bs[2][4 * (CHUNK_K / 32) * 512];

  const int lane = threadIdx.x & 31;
  const int wave = threadIdx.x >> 5;
  const int m0   = blockIdx.x * 256 + wave * 32;
  const int n0   = blockIdx.y * 64;
  const int lrow = lane & 15;
  const int lhi  = lane >> 4;

  const unsigned lds_base = (unsigned)(uintptr_t)(void*)&Bs[0][0];
  const int nchunks = (K + CHUNK_K - 1) / CHUNK_K;

  v8f acc[2][4] = {};

  const bf16_t* Ap0 = A + (size_t)(m0 + lrow) * K + lhi * 16;
  const bf16_t* Ap1 = Ap0 + (size_t)16 * K;

  // stage chunk 0 into buffer 0
  {
    int kc = K < CHUNK_K ? K : CHUNK_K;
    int kcnt = kc >> 5;
    if (threadIdx.x < 32) {
      for (int j = 0; j < 4; j++)
        tdm_stage_B(W + (size_t)(n0 + j * 16) * K, lds_base + (unsigned)(j * kcnt) * 1024u, K, kcnt);
      __builtin_amdgcn_s_wait_tensorcnt(0);
    }
  }
  __syncthreads();

  for (int c = 0; c < nchunks; c++) {
    const int kc   = (K - c * CHUNK_K) < CHUNK_K ? (K - c * CHUNK_K) : CHUNK_K;
    const int kcnt = kc >> 5;

    // kick off TDM for next chunk into the other buffer
    if (c + 1 < nchunks) {
      int kc2 = (K - (c + 1) * CHUNK_K) < CHUNK_K ? (K - (c + 1) * CHUNK_K) : CHUNK_K;
      int kcnt2 = kc2 >> 5;
      if (threadIdx.x < 32) {
        unsigned bufoff = lds_base + (unsigned)(((c + 1) & 1) ? 32768u : 0u);
        for (int j = 0; j < 4; j++)
          tdm_stage_B(W + (size_t)(n0 + j * 16) * K + (size_t)(c + 1) * CHUNK_K,
                      bufoff + (unsigned)(j * kcnt2) * 1024u, K, kcnt2);
      }
    }

    const bf16_t* BsBase = &Bs[c & 1][0];
    for (int ks = 0; ks < kcnt; ks++) {
      const int k0 = c * CHUNK_K + ks * 32;

      v16bf af[2];
      {
        v8bf lo0 = *(const v8bf*)(Ap0 + k0);
        v8bf hi0 = *(const v8bf*)(Ap0 + k0 + 8);
        af[0] = concat8(lo0, hi0);
        v8bf lo1 = *(const v8bf*)(Ap1 + k0);
        v8bf hi1 = *(const v8bf*)(Ap1 + k0 + 8);
        af[1] = concat8(lo1, hi1);
      }

      v16bf bfm[4];
      #pragma unroll
      for (int j = 0; j < 4; j++) {
        const bf16_t* p = BsBase + (size_t)(j * kcnt + ks) * 512 + lane * 8;
        v8bf lo = *(const v8bf*)p;          // frag + L*16B
        v8bf hi = *(const v8bf*)(p + 256);  // frag + 512B + L*16B
        bfm[j] = concat8(lo, hi);
      }

      #pragma unroll
      for (int i = 0; i < 2; i++)
        #pragma unroll
        for (int j = 0; j < 4; j++)
          acc[i][j] = __builtin_amdgcn_wmma_f32_16x16x32_bf16(
              false, af[i], false, bfm[j], (short)0, acc[i][j], false, false);
    }

    if (c + 1 < nchunks) {
      if (threadIdx.x < 32) __builtin_amdgcn_s_wait_tensorcnt(0);
      __syncthreads();
    }
  }

  #pragma unroll
  for (int i = 0; i < 2; i++) {
    const int rbase = m0 + i * 16 + lhi * 8;
    #pragma unroll
    for (int j = 0; j < 4; j++) {
      const int col = n0 + j * 16 + lrow;
      const float bb = bias ? bias[col] : 0.0f;
      #pragma unroll
      for (int e = 0; e < 8; e++) {
        float v = acc[i][j][e] + bb;
        if (relu) v = fmaxf(v, 0.0f);
        const size_t idx = (size_t)(rbase + e) * N + col;
        if (Cf) Cf[idx] = v;
        if (Cb) Cb[idx] = (bf16_t)v;
      }
    }
  }
  (void)M;
}

// ------------------------------------------------------------------
// Banded causal attention, window [i-4, i]; one wave per (s, b, head).
// qkv: bf16 [S*B, 3H] (q|k|v), ctx: bf16 [S*B, H]. dh=256 -> 8 f32/lane.
// ------------------------------------------------------------------
__global__ __launch_bounds__(128) void k_attn(const bf16_t* __restrict__ qkv,
                                              bf16_t* __restrict__ ctx)
{
  const int gw   = blockIdx.x * 4 + (threadIdx.x >> 5);
  const int lane = threadIdx.x & 31;
  const int h = gw & (NHEAD_ - 1);
  const int b = (gw >> 2) & (B_ - 1);
  const int s = gw >> 4;

  const size_t d_off = (size_t)h * DH_ + lane * 8;
  const v8bf qv = *(const v8bf*)(qkv + ((size_t)s * B_ + b) * (3 * H_) + d_off);
  float q[8];
  #pragma unroll
  for (int e = 0; e < 8; e++) q[e] = (float)qv[e];

  int j0 = s - WINDOW_; if (j0 < 0) j0 = 0;
  const int nj = s - j0 + 1;

  float sc[WINDOW_ + 1];
  float mx = -1e30f;
  for (int jj = 0; jj < nj; jj++) {
    const size_t rk = ((size_t)(j0 + jj) * B_ + b) * (3 * H_) + H_ + d_off;
    const v8bf kv = *(const v8bf*)(qkv + rk);
    float d = 0.0f;
    #pragma unroll
    for (int e = 0; e < 8; e++) d += q[e] * (float)kv[e];
    #pragma unroll
    for (int off = 16; off > 0; off >>= 1) d += __shfl_xor(d, off, 32);
    d *= 0.0625f;                 // 1/sqrt(256)
    sc[jj] = d;
    mx = fmaxf(mx, d);
  }

  float p[WINDOW_ + 1];
  float sum = 0.0f;
  for (int jj = 0; jj < nj; jj++) { p[jj] = __expf(sc[jj] - mx); sum += p[jj]; }
  const float inv = 1.0f / sum;

  float o[8] = {0.f, 0.f, 0.f, 0.f, 0.f, 0.f, 0.f, 0.f};
  for (int jj = 0; jj < nj; jj++) {
    const size_t rv = ((size_t)(j0 + jj) * B_ + b) * (3 * H_) + 2 * H_ + d_off;
    const v8bf vv = *(const v8bf*)(qkv + rv);
    #pragma unroll
    for (int e = 0; e < 8; e++) o[e] += p[jj] * (float)vv[e];
  }

  v8bf ov;
  #pragma unroll
  for (int e = 0; e < 8; e++) ov[e] = (bf16_t)(o[e] * inv);
  *(v8bf*)(ctx + ((size_t)s * B_ + b) * H_ + d_off) = ov;
}

// ------------------------------------------------------------------
// x = LayerNorm(x + y) * gamma + beta; emits f32 (residual master)
// and bf16 (next GEMM input). One 256-thread block per row (H=1024).
// ------------------------------------------------------------------
__global__ __launch_bounds__(256) void k_add_ln(
    const float* __restrict__ x, const float* __restrict__ y,
    const float* __restrict__ gam, const float* __restrict__ bet,
    float* __restrict__ xo, bf16_t* __restrict__ xb)
{
  __shared__ float r1[256], r2[256];
  const int row = blockIdx.x, tid = threadIdx.x;
  const float* xr = x + (size_t)row * H_;
  const float* yr = y + (size_t)row * H_;

  float v[4], s = 0.0f, ss = 0.0f;
  #pragma unroll
  for (int i = 0; i < 4; i++) {
    const int c = tid + i * 256;
    v[i] = xr[c] + yr[c];
    s += v[i]; ss += v[i] * v[i];
  }
  r1[tid] = s; r2[tid] = ss;
  __syncthreads();
  for (int st = 128; st > 0; st >>= 1) {
    if (tid < st) { r1[tid] += r1[tid + st]; r2[tid] += r2[tid + st]; }
    __syncthreads();
  }
  const float mean = r1[0] * (1.0f / H_);
  const float var  = r2[0] * (1.0f / H_) - mean * mean;
  const float rs   = rsqrtf(var + 1e-5f);

  #pragma unroll
  for (int i = 0; i < 4; i++) {
    const int c = tid + i * 256;
    const float o = (v[i] - mean) * rs * gam[c] + bet[c];
    xo[(size_t)row * H_ + c] = o;
    xb[(size_t)row * H_ + c] = (bf16_t)o;
  }
}

// ------------------------------------------------------------------
extern "C" void kernel_launch(void* const* d_in, const int* in_sizes, int n_in,
                              void* d_out, int out_size, void* d_ws, size_t ws_size,
                              hipStream_t stream)
{
  (void)in_sizes; (void)n_in; (void)out_size; (void)ws_size;

  const float* src  = (const float*)d_in[0];
  const float* Wenc = (const float*)d_in[1];
  const float* benc = (const float*)d_in[2];
  const float* Wqkv = (const float*)d_in[3];
  const float* bqkv = (const float*)d_in[4];
  const float* Wo   = (const float*)d_in[5];
  const float* bo   = (const float*)d_in[6];
  const float* W1   = (const float*)d_in[7];
  const float* b1   = (const float*)d_in[8];
  const float* W2   = (const float*)d_in[9];
  const float* b2   = (const float*)d_in[10];
  const float* ln1s = (const float*)d_in[11];
  const float* ln1b = (const float*)d_in[12];
  const float* ln2s = (const float*)d_in[13];
  const float* ln2b = (const float*)d_in[14];
  const float* Wout = (const float*)d_in[15];
  const float* bout = (const float*)d_in[16];
  float* out = (float*)d_out;

  const size_t M = (size_t)S_ * B_;  // 8192 token rows

  // workspace partition
  char* base = (char*)d_ws;
  size_t off = 0;
  auto alloc = [&](size_t bytes) -> char* {
    char* p = base + off;
    off += (bytes + 255) & ~(size_t)255;
    return p;
  };
  float*  x_f    = (float*) alloc(M * H_ * 4);
  bf16_t* x_b    = (bf16_t*)alloc(M * H_ * 2);
  bf16_t* qkv_b  = (bf16_t*)alloc(M * 3 * H_ * 2);
  bf16_t* ctx_b  = (bf16_t*)alloc(M * H_ * 2);
  bf16_t* h1_b   = (bf16_t*)alloc(M * DFF_ * 2);
  float*  tmp_f  = (float*) alloc(M * H_ * 4);
  bf16_t* src_b  = (bf16_t*)alloc(M * IN_ * 2);
  bf16_t* Wenc_b = (bf16_t*)alloc((size_t)H_ * IN_ * 2);
  bf16_t* Wqkv_b = (bf16_t*)alloc((size_t)L_ * 3 * H_ * H_ * 2);
  bf16_t* Wo_b   = (bf16_t*)alloc((size_t)L_ * H_ * H_ * 2);
  bf16_t* W1_b   = (bf16_t*)alloc((size_t)L_ * DFF_ * H_ * 2);
  bf16_t* W2_b   = (bf16_t*)alloc((size_t)L_ * H_ * DFF_ * 2);
  bf16_t* Wout_b = (bf16_t*)alloc((size_t)OUT_ * H_ * 2);

  auto cvt = [&](const float* in, bf16_t* o, size_t n) {
    int blocks = (int)((n + 255) / 256);
    if (blocks > 16384) blocks = 16384;
    k_cvt_bf16<<<blocks, 256, 0, stream>>>(in, o, n);
  };

  cvt(src,  src_b,  M * IN_);
  cvt(Wenc, Wenc_b, (size_t)H_ * IN_);
  cvt(Wqkv, Wqkv_b, (size_t)L_ * 3 * H_ * H_);
  cvt(Wo,   Wo_b,   (size_t)L_ * H_ * H_);
  cvt(W1,   W1_b,   (size_t)L_ * DFF_ * H_);
  cvt(W2,   W2_b,   (size_t)L_ * H_ * DFF_);
  cvt(Wout, Wout_b, (size_t)OUT_ * H_);

  const dim3 blk(256);
  const unsigned mb = (unsigned)(M / 256);

  // encoder: x = src @ Wenc^T + benc  (emit f32 + bf16)
  k_gemm_bf16<<<dim3(mb, H_ / 64), blk, 0, stream>>>(
      src_b, Wenc_b, benc, x_f, x_b, (int)M, H_, IN_, 0);

  for (int l = 0; l < L_; l++) {
    // qkv = x @ Wqkv^T + bqkv  (bf16 only)
    k_gemm_bf16<<<dim3(mb, 3 * H_ / 64), blk, 0, stream>>>(
        x_b, Wqkv_b + (size_t)l * 3 * H_ * H_, bqkv + (size_t)l * 3 * H_,
        nullptr, qkv_b, (int)M, 3 * H_, H_, 0);

    // banded attention -> ctx (bf16)
    k_attn<<<(S_ * B_ * NHEAD_) / 4, 128, 0, stream>>>(qkv_b, ctx_b);

    // attn_out = ctx @ Wo^T + bo  (f32)
    k_gemm_bf16<<<dim3(mb, H_ / 64), blk, 0, stream>>>(
        ctx_b, Wo_b + (size_t)l * H_ * H_, bo + (size_t)l * H_,
        tmp_f, nullptr, (int)M, H_, H_, 0);

    // x = LN(x + attn_out)
    k_add_ln<<<(unsigned)M, blk, 0, stream>>>(
        x_f, tmp_f, ln1s + (size_t)l * H_, ln1b + (size_t)l * H_, x_f, x_b);

    // h = relu(x @ W1^T + b1)  (bf16 only)
    k_gemm_bf16<<<dim3(mb, DFF_ / 64), blk, 0, stream>>>(
        x_b, W1_b + (size_t)l * DFF_ * H_, b1 + (size_t)l * DFF_,
        nullptr, h1_b, (int)M, DFF_, H_, 1);

    // ff = h @ W2^T + b2  (f32)
    k_gemm_bf16<<<dim3(mb, H_ / 64), blk, 0, stream>>>(
        h1_b, W2_b + (size_t)l * H_ * DFF_, b2 + (size_t)l * H_,
        tmp_f, nullptr, (int)M, H_, DFF_, 0);

    // x = LN(x + ff)
    k_add_ln<<<(unsigned)M, blk, 0, stream>>>(
        x_f, tmp_f, ln2s + (size_t)l * H_, ln2b + (size_t)l * H_, x_f, x_b);
  }

  // out = x @ Wout^T + bout  (f32 -> d_out)
  k_gemm_bf16<<<dim3(mb, OUT_ / 64), blk, 0, stream>>>(
      x_b, Wout_b, bout, out, nullptr, (int)M, OUT_, H_, 0);
}
